// MigraineMamba_8100308320553
// MI455X (gfx1250) — matrix operand order repaired and checked by reading, not compiled, not run on hardware
//
#include <hip/hip_runtime.h>
#include <hip/hip_bf16.h>

typedef _Float16 h16;
typedef _Float16 v8h  __attribute__((ext_vector_type(8)));
typedef _Float16 v16h __attribute__((ext_vector_type(16)));
typedef float    v8f  __attribute__((ext_vector_type(8)));

#define B_SZ   32768
#define LSEQ   14
#define DM     64
#define DI     128
#define DST    16
#define NB     8            // batch elements per workgroup
#define MROWS  (NB * LSEQ)  // 112 GEMM rows
#define MTILES (MROWS / 16) // 7
#define NTH    256

// ---- LDS arena layout (bytes) ----
#define OFF_X     0        // h16 [112][64]   residual stream x
#define OFF_XN    14336    // h16 [112][64]   ln(x) / xdbl[112][48] / head scratch (f32)
#define OFF_XZ    28672    // h16 [112][256]  xz  / stage0 f32 scratch / Q,K,V
#define OFF_XI    86016    // h16 [112][128]  xi -> y -> attn O
#define OFF_W     114688   // h16 weight staging (transposed, max 256x64) / f32 A[128][16]
#define OFF_STATS 147456   // f32 LN stats (448 floats)
#define SMEM_BYTES 149504

struct PtrPack { const void* p[67]; };
#define PF(i) ((const float*)pp.p[(i)])
#define PI(i) ((const int*)  pp.p[(i)])

__device__ __forceinline__ float sigmoidf_(float x) { return 1.f / (1.f + __expf(-x)); }
__device__ __forceinline__ float siluf_(float x)    { return x * sigmoidf_(x); }
__device__ __forceinline__ float geluf_(float x) {
    float x3 = x * x * x;
    return 0.5f * x * (1.f + tanhf(0.7978845608028654f * (x + 0.044715f * x3)));
}
__device__ __forceinline__ float softplusf_(float x) { return (x > 20.f) ? x : log1pf(__expf(x)); }

// One 16x16 D-tile. A[M][lda] row-major f16 in LDS. Bt[N][ldk] = B^T (N-major) f16 in LDS.
// K = ksteps*32 via v_wmma_f32_16x16x32_f16. Per-lane operands are contiguous 16B runs
// so each operand fragment is exactly two ds_load_b128.
__device__ __forceinline__ v8f wmma_mm(const h16* As, int lda, const h16* Bt, int ldk,
                                       int m0, int n0, int ksteps, v8f acc) {
    const int lane = threadIdx.x & 31;
    const int lg   = lane >> 4;
    const int mr   = m0 + (lane & 15);
    const int nc   = n0 + (lane & 15);
    for (int ks = 0; ks < ksteps; ++ks) {
        // A: elements 0..7 -> K = ks*32 + lg*8 + e ; elements 8..15 -> +16
        const h16* ap = As + mr * lda + ks * 32 + (lg << 3);
        v8h a0 = *(const v8h*)(ap);
        v8h a1 = *(const v8h*)(ap + 16);
        // B^T: elements 0..7 -> K = ks*32 + lg*8 + e ; elements 8..15 -> +16, at fixed column nc
        const h16* bp = Bt + nc * ldk + ks * 32 + (lg << 3);
        v8h b0 = *(const v8h*)(bp);
        v8h b1 = *(const v8h*)(bp + 16);
        v16h a = __builtin_shufflevector(a0, a1, 0,1,2,3,4,5,6,7,8,9,10,11,12,13,14,15);
        v16h b = __builtin_shufflevector(b0, b1, 0,1,2,3,4,5,6,7,8,9,10,11,12,13,14,15);
        acc = __builtin_amdgcn_wmma_f32_16x16x32_f16(false, a, false, b, (short)0, acc,
                                                     false, false);
    }
    return acc;
}

__global__ __launch_bounds__(NTH, 1)
void migraine_fused(PtrPack pp, float* __restrict__ out) {
    __shared__ __align__(16) char smem[SMEM_BYTES];
    h16*   sX    = (h16*)(smem + OFF_X);
    h16*   sXn   = (h16*)(smem + OFF_XN);
    h16*   sXZ   = (h16*)(smem + OFF_XZ);
    h16*   sXi   = (h16*)(smem + OFF_XI);
    h16*   sW    = (h16*)(smem + OFF_W);      // transposed weights
    float* sA    = (float*)(smem + OFF_W);    // scan-time alias of sW
    h16*   sXdbl = (h16*)(smem + OFF_XN);     // [112][48] alias of sXn
    float* sFpre = (float*)(smem + OFF_XZ);   // [112][48] stage-0 alias
    float* sScr  = (float*)(smem + OFF_XN);   // head-stage alias
    float* sStats= (float*)(smem + OFF_STATS);

    const int tid  = threadIdx.x;
    const int lane = tid & 31;
    const int wave = tid >> 5;
    const int lg   = lane >> 4;
    const int nlo  = lane & 15;
    const int m0   = wave * 16;
    const int b0   = blockIdx.x * NB;

    // ---------------- Stage 0: feature embedding ----------------
    // 0a: pre-LN linear features (cont 32 | bin 16) -> f32 scratch
    for (int idx = tid; idx < MROWS * 48; idx += NTH) {
        int r = idx / 48, j = idx % 48;
        int b = b0 + r / LSEQ, l = r % LSEQ;
        float acc;
        if (j < 32) {
            acc = PF(20)[j];                       // cont_b
            const float* w  = PF(23);              // cont_w (8,32)
            const float* cv = PF(0) + ((size_t)b * LSEQ + l) * 8;
#pragma unroll
            for (int i = 0; i < 8; ++i) acc += cv[i] * w[i * 32 + j];
        } else {
            int jj = j - 32;
            acc = PF(16)[jj];                      // bin_b
            const float* w  = PF(19);              // bin_w (8,16)
            const float* bv = PF(1) + ((size_t)b * LSEQ + l) * 8;
#pragma unroll
            for (int i = 0; i < 8; ++i) acc += bv[i] * w[i * 16 + jj];
        }
        sFpre[r * 48 + j] = acc;
    }
    __syncthreads();
    // 0b: LN stats per (row, segment)
    for (int t = tid; t < MROWS * 2; t += NTH) {
        int r = t >> 1, seg = t & 1;
        int o = seg ? 32 : 0, n = seg ? 16 : 32;
        float m = 0.f;
        for (int j = 0; j < n; ++j) m += sFpre[r * 48 + o + j];
        m /= (float)n;
        float v = 0.f;
        for (int j = 0; j < n; ++j) { float d = sFpre[r * 48 + o + j] - m; v += d * d; }
        v /= (float)n;
        sStats[t * 2]     = m;
        sStats[t * 2 + 1] = rsqrtf(v + 1e-5f);
    }
    __syncthreads();
    // 0c: LN + gelu + embedding gathers -> concat(64) f16
    for (int idx = tid; idx < MROWS * 64; idx += NTH) {
        int r = idx >> 6, c = idx & 63;
        int b = b0 + r / LSEQ, l = r % LSEQ;
        float val;
        if (c < 32) {
            float mu = sStats[(r * 2) * 2], rs = sStats[(r * 2) * 2 + 1];
            val = geluf_((sFpre[r * 48 + c] - mu) * rs * PF(22)[c] + PF(21)[c]);
        } else if (c < 48) {
            int jj = c - 32;
            float mu = sStats[(r * 2 + 1) * 2], rs = sStats[(r * 2 + 1) * 2 + 1];
            val = geluf_((sFpre[r * 48 + 32 + jj] - mu) * rs * PF(18)[jj] + PF(17)[jj]);
        } else if (c < 56) {
            int m = PI(2)[(size_t)b * LSEQ + l];
            m = (m < 0) ? 29 : m; m = (m > 29) ? 29 : m;
            val = PF(25)[m * 8 + (c - 48)];
        } else {
            int d = PI(3)[(size_t)b * LSEQ + l];
            val = PF(24)[d * 8 + (c - 56)];
        }
        sXn[r * 64 + c] = (h16)val;
    }
    __syncthreads();
    // 0d: project to d_model with WMMA, add bias + positional.  sW[n*64+k] = proj_w[k*64+n]
    for (int idx = tid; idx < 64 * 64; idx += NTH) {
        int n = idx >> 6, k = idx & 63;
        sW[idx] = (h16)PF(27)[k * 64 + n];
    }
    __syncthreads();
    if (wave < MTILES) {
        for (int nt = 0; nt < 4; ++nt) {
            v8f acc = {0.f, 0.f, 0.f, 0.f, 0.f, 0.f, 0.f, 0.f};
            acc = wmma_mm(sXn, 64, sW, 64, m0, nt * 16, 2, acc);
            int n = nt * 16 + nlo;
#pragma unroll
            for (int e = 0; e < 8; ++e) {
                int r = m0 + e + (lg << 3);
                int l = r % LSEQ;
                sX[r * 64 + n] = (h16)(acc[e] + PF(26)[n] + PF(66)[l * 64 + n]);
            }
        }
    }
    __syncthreads();

    // ---------------- Mamba layers ----------------
    for (int layer = 0; layer < 2; ++layer) {
        const int lp = 40 + layer * 13;
        // LN(x) -> sXn
        for (int r = tid; r < MROWS; r += NTH) {
            float m = 0.f;
            for (int c = 0; c < 64; ++c) m += (float)sX[r * 64 + c];
            m *= (1.f / 64.f);
            float v = 0.f;
            for (int c = 0; c < 64; ++c) { float d = (float)sX[r * 64 + c] - m; v += d * d; }
            float rs = rsqrtf(v * (1.f / 64.f) + 1e-5f);
            for (int c = 0; c < 64; ++c)
                sXn[r * 64 + c] = (h16)(((float)sX[r * 64 + c] - m) * rs * PF(lp + 9)[c] + PF(lp + 8)[c]);
        }
        __syncthreads();
        // in_proj: (112x64)@(64x256) -> xz.  sW[n*64+k] = in_w[k*256+n]
        for (int idx = tid; idx < 64 * 256; idx += NTH) {
            int n = idx >> 6, k = idx & 63;
            sW[idx] = (h16)PF(lp + 7)[k * 256 + n];
        }
        __syncthreads();
        if (wave < MTILES) {
            for (int nt = 0; nt < 16; ++nt) {
                v8f acc = {0.f, 0.f, 0.f, 0.f, 0.f, 0.f, 0.f, 0.f};
                acc = wmma_mm(sXn, 64, sW, 64, m0, nt * 16, 2, acc);
                int n = nt * 16 + nlo;
#pragma unroll
                for (int e = 0; e < 8; ++e) {
                    int r = m0 + e + (lg << 3);
                    sXZ[r * 256 + n] = (h16)(acc[e] + PF(lp + 6)[n]);
                }
            }
        }
        __syncthreads();
        // depthwise causal conv + silu -> xi
        for (int idx = tid; idx < MROWS * DI; idx += NTH) {
            int r = idx >> 7, d = idx & 127;
            int bb = r / LSEQ, l = r % LSEQ;
            float acc = PF(lp + 2)[d];
            const float* cw = PF(lp + 3) + d * 4;
#pragma unroll
            for (int t = 0; t < 4; ++t) {
                int ll = l - 3 + t;
                if (ll >= 0) acc += cw[t] * (float)sXZ[(bb * LSEQ + ll) * 256 + d];
            }
            sXi[idx] = (h16)siluf_(acc);
        }
        __syncthreads();
        // x_proj: (112x128)@(128x36->48) -> xdbl.  sW[n*128+k] = xproj_w[k*36+n] (pad n>=36)
        for (int idx = tid; idx < 48 * 128; idx += NTH) {
            int n = idx >> 7, k = idx & 127;
            sW[idx] = (h16)((n < 36) ? PF(lp + 12)[k * 36 + n] : 0.f);
        }
        __syncthreads();
        if (wave < MTILES) {
            for (int nt = 0; nt < 3; ++nt) {
                v8f acc = {0.f, 0.f, 0.f, 0.f, 0.f, 0.f, 0.f, 0.f};
                acc = wmma_mm(sXi, 128, sW, 128, m0, nt * 16, 4, acc);
                int n = nt * 16 + nlo;
#pragma unroll
                for (int e = 0; e < 8; ++e) {
                    int r = m0 + e + (lg << 3);
                    sXdbl[r * 48 + n] = (h16)acc[e];
                }
            }
        }
        __syncthreads();
        // A = -exp(A_log) staged into LDS (aliases sW)
        for (int idx = tid; idx < DI * DST; idx += NTH) sA[idx] = -__expf(PF(lp + 0)[idx]);
        __syncthreads();
        // selective scan: each thread owns (b,d) lanes across all 14 steps
        for (int pair = tid; pair < NB * DI; pair += NTH) {
            int bb = pair >> 7, d = pair & 127;
            const float dtw0 = PF(lp + 5)[0 * 128 + d], dtw1 = PF(lp + 5)[1 * 128 + d];
            const float dtw2 = PF(lp + 5)[2 * 128 + d], dtw3 = PF(lp + 5)[3 * 128 + d];
            const float dtb = PF(lp + 4)[d], Dd = PF(lp + 1)[d];
            const float* Ad = sA + d * 16;
            float h[16];
#pragma unroll
            for (int n = 0; n < 16; ++n) h[n] = 0.f;
            for (int l = 0; l < LSEQ; ++l) {
                int r = bb * LSEQ + l;
                const h16* xd = sXdbl + r * 48;
                float u  = (float)sXi[r * 128 + d];
                float dt = softplusf_((float)xd[0] * dtw0 + (float)xd[1] * dtw1 +
                                      (float)xd[2] * dtw2 + (float)xd[3] * dtw3 + dtb);
                float y = 0.f;
#pragma unroll
                for (int n = 0; n < 16; ++n) {
                    float dA = __expf(dt * Ad[n]);
                    h[n] = h[n] * dA + dt * u * (float)xd[4 + n];
                    y += h[n] * (float)xd[20 + n];
                }
                float z = (float)sXZ[r * 256 + 128 + d];
                sXi[r * 128 + d] = (h16)((y + u * Dd) * siluf_(z));
            }
        }
        __syncthreads();
        // out_proj: (112x128)@(128x64), residual add.  sW[n*128+k] = out_w[k*64+n]
        for (int idx = tid; idx < 64 * 128; idx += NTH) {
            int n = idx >> 7, k = idx & 127;
            sW[idx] = (h16)PF(lp + 11)[k * 64 + n];
        }
        __syncthreads();
        if (wave < MTILES) {
            for (int nt = 0; nt < 4; ++nt) {
                v8f acc = {0.f, 0.f, 0.f, 0.f, 0.f, 0.f, 0.f, 0.f};
                acc = wmma_mm(sXi, 128, sW, 128, m0, nt * 16, 4, acc);
                int n = nt * 16 + nlo;
#pragma unroll
                for (int e = 0; e < 8; ++e) {
                    int r = m0 + e + (lg << 3);
                    sX[r * 64 + n] = (h16)((float)sX[r * 64 + n] + acc[e] + PF(lp + 10)[n]);
                }
            }
        }
        __syncthreads();
    }

    // ---------------- Multi-head attention (residual) ----------------
    h16* sQ = sXZ;
    h16* sK = sXZ + MROWS * 64;
    h16* sV = sXZ + 2 * MROWS * 64;
    h16* sO = sXi;
    // stage q/k/v/o weights transposed: sW[slot*4096 + n*64 + k] = w[k*64+n]
    for (int idx = tid; idx < 4096; idx += NTH) {
        int n = idx >> 6, k = idx & 63;
        sW[idx]         = (h16)PF(13)[k * 64 + n];   // q_w^T
        sW[4096 + idx]  = (h16)PF(9)[k * 64 + n];    // k_w^T
        sW[8192 + idx]  = (h16)PF(15)[k * 64 + n];   // v_w^T
        sW[12288 + idx] = (h16)PF(11)[k * 64 + n];   // o_w^T
    }
    __syncthreads();
    if (wave < MTILES) {
        for (int nt = 0; nt < 4; ++nt) {
            v8f aq = {0.f,0.f,0.f,0.f,0.f,0.f,0.f,0.f};
            v8f ak = {0.f,0.f,0.f,0.f,0.f,0.f,0.f,0.f};
            v8f av = {0.f,0.f,0.f,0.f,0.f,0.f,0.f,0.f};
            aq = wmma_mm(sX, 64, sW,         64, m0, nt * 16, 2, aq);
            ak = wmma_mm(sX, 64, sW + 4096,  64, m0, nt * 16, 2, ak);
            av = wmma_mm(sX, 64, sW + 8192,  64, m0, nt * 16, 2, av);
            int n = nt * 16 + nlo;
#pragma unroll
            for (int e = 0; e < 8; ++e) {
                int r = m0 + e + (lg << 3);
                sQ[r * 64 + n] = (h16)(aq[e] + PF(12)[n]);
                sK[r * 64 + n] = (h16)(ak[e] + PF(8)[n]);
                sV[r * 64 + n] = (h16)(av[e] + PF(14)[n]);
            }
        }
    }
    __syncthreads();
    // per (b, head, q) softmax attention, L=14, d_head=16
    for (int idx = tid; idx < NB * 4 * LSEQ; idx += NTH) {
        int q = idx % LSEQ, hh = (idx / LSEQ) & 3, bb = idx / (LSEQ * 4);
        const h16* Q = sQ + (bb * LSEQ + q) * 64 + hh * 16;
        float sc[LSEQ], mx = -1e30f;
        for (int k = 0; k < LSEQ; ++k) {
            const h16* K = sK + (bb * LSEQ + k) * 64 + hh * 16;
            float s = 0.f;
#pragma unroll
            for (int d = 0; d < 16; ++d) s += (float)Q[d] * (float)K[d];
            s *= 0.25f;
            sc[k] = s; mx = fmaxf(mx, s);
        }
        float den = 0.f;
        for (int k = 0; k < LSEQ; ++k) { sc[k] = __expf(sc[k] - mx); den += sc[k]; }
        float inv = 1.f / den;
#pragma unroll
        for (int d = 0; d < 16; ++d) {
            float o = 0.f;
            for (int k = 0; k < LSEQ; ++k)
                o += sc[k] * (float)sV[(bb * LSEQ + k) * 64 + hh * 16 + d];
            sO[(bb * LSEQ + q) * 64 + hh * 16 + d] = (h16)(o * inv);
        }
    }
    __syncthreads();
    // O projection, residual add
    if (wave < MTILES) {
        for (int nt = 0; nt < 4; ++nt) {
            v8f acc = {0.f,0.f,0.f,0.f,0.f,0.f,0.f,0.f};
            acc = wmma_mm(sO, 64, sW + 12288, 64, m0, nt * 16, 2, acc);
            int n = nt * 16 + nlo;
#pragma unroll
            for (int e = 0; e < 8; ++e) {
                int r = m0 + e + (lg << 3);
                sX[r * 64 + n] = (h16)((float)sX[r * 64 + n] + acc[e] + PF(10)[n]);
            }
        }
    }
    __syncthreads();

    // ---------------- Aggregation + knowledge + head ----------------
    float* scScore = sScr;          // [112]
    float* scWn    = sScr + 112;    // [112]
    float* scAgg   = sScr + 224;    // [8][64]
    float* scC1    = sScr + 736;    // [8][32]
    float* scClin  = sScr + 992;    // [8][32]
    float* scComb  = sScr + 1248;   // [8][64]
    float* scH1    = sScr + 1760;   // [8][64]
    float* scH2    = sScr + 2272;   // [8][32]

    for (int r = tid; r < MROWS; r += NTH) {
        float s = PF(5)[0];
        for (int d = 0; d < 64; ++d) s += (float)sX[r * 64 + d] * PF(7)[d];
        scScore[r] = s;
    }
    __syncthreads();
    for (int bb = tid; bb < NB; bb += NTH) {
        float mx = -1e30f;
        for (int l = 0; l < LSEQ; ++l) mx = fmaxf(mx, scScore[bb * LSEQ + l]);
        float den = 0.f, w[LSEQ];
        for (int l = 0; l < LSEQ; ++l) { w[l] = __expf(scScore[bb * LSEQ + l] - mx); den += w[l]; }
        float wsum = 0.f;
        for (int l = 0; l < LSEQ; ++l) { w[l] = (w[l] / den) * PF(6)[l]; wsum += w[l]; }
        float inv = 1.f / wsum;
        for (int l = 0; l < LSEQ; ++l) scWn[bb * LSEQ + l] = w[l] * inv;
    }
    __syncthreads();
    for (int idx = tid; idx < NB * 64; idx += NTH) {
        int bb = idx >> 6, d = idx & 63;
        float a = 0.f;
        for (int l = 0; l < LSEQ; ++l)
            a += scWn[bb * LSEQ + l] * (float)sX[(bb * LSEQ + l) * 64 + d];
        scAgg[idx] = a;
    }
    for (int idx = tid; idx < NB * 32; idx += NTH) {
        int bb = idx >> 5, j = idx & 31;
        float a = PF(34)[j];
        const float* cv = PF(4) + (size_t)(b0 + bb) * 8;
#pragma unroll
        for (int i = 0; i < 8; ++i) a += cv[i] * PF(38)[i * 32 + j];
        scC1[idx] = fmaxf(a, 0.f);
    }
    __syncthreads();
    for (int idx = tid; idx < NB * 32; idx += NTH) {
        int bb = idx >> 5, j = idx & 31;
        float a = PF(35)[j];
        for (int k = 0; k < 32; ++k) a += scC1[bb * 32 + k] * PF(39)[k * 32 + j];
        scClin[idx] = a;
    }
    __syncthreads();
    for (int idx = tid; idx < NB * 64; idx += NTH) {
        int bb = idx >> 6, c = idx & 63;
        float a = PF(36)[c];
        for (int k = 0; k < 64; ++k) a += scAgg[bb * 64 + k] * PF(37)[k * 64 + c];
        for (int k = 0; k < 32; ++k) a += scClin[bb * 32 + k] * PF(37)[(64 + k) * 64 + c];
        scComb[idx] = a;
    }
    __syncthreads();
    for (int idx = tid; idx < NB * 64; idx += NTH) {
        int bb = idx >> 6, c = idx & 63;
        float a = PF(28)[c];
        for (int k = 0; k < 64; ++k) a += scComb[bb * 64 + k] * PF(31)[k * 64 + c];
        scH1[idx] = fmaxf(a, 0.f);
    }
    __syncthreads();
    for (int idx = tid; idx < NB * 32; idx += NTH) {
        int bb = idx >> 5, c = idx & 31;
        float a = PF(29)[c];
        for (int k = 0; k < 64; ++k) a += scH1[bb * 64 + k] * PF(32)[k * 32 + c];
        scH2[idx] = fmaxf(a, 0.f);
    }
    __syncthreads();
    for (int bb = tid; bb < NB; bb += NTH) {
        float a = PF(30)[0];
        for (int k = 0; k < 32; ++k) a += scH2[bb * 32 + k] * PF(33)[k];
        out[b0 + bb] = a;
    }
}

extern "C" void kernel_launch(void* const* d_in, const int* in_sizes, int n_in,
                              void* d_out, int out_size, void* d_ws, size_t ws_size,
                              hipStream_t stream) {
    (void)in_sizes; (void)out_size; (void)d_ws; (void)ws_size;
    PtrPack pp;
    for (int i = 0; i < 67; ++i) pp.p[i] = (i < n_in) ? d_in[i] : nullptr;
    dim3 grid(B_SZ / NB), block(NTH);
    hipLaunchKernelGGL(migraine_fused, grid, block, 0, stream, pp, (float*)d_out);
}
